// KAN_27547920237095
// MI455X (gfx1250) — matrix operand order repaired and checked by reading, not compiled, not run on hardware
//
#include <hip/hip_runtime.h>
#include <hip/hip_bf16.h>

// ---------------------------------------------------------------------------
// KAN 3-layer forward as 3 bf16 WMMA GEMMs (f32 accumulate).
//   X_aug[n, f*9+j] = { relu(x), B0..B7(x) }      (bf16)
//   W_aug[o, f*9+j] = { base_w, spline_w*scaler } (bf16)
//   H = X_aug @ W_augT                            (f32)
// GEMM: 128x256 block tile, 8 waves @ 64x64 (16 WMMA frags each),
// triple-buffered LDS fed by GLOBAL_LOAD_ASYNC_TO_LDS_B128 (ASYNCcnt).
// ---------------------------------------------------------------------------

#define BATCH  4096
#define POINTS 512
#define EMBD   2048
#define F0     1023
#define F0P    1024
#define K0     (F0P * 9)   // 9216
#define K1     (EMBD * 9)  // 18432

typedef __attribute__((ext_vector_type(16))) __bf16 bf16x16;
typedef __attribute__((ext_vector_type(8)))  __bf16 bf16x8;
typedef __attribute__((ext_vector_type(8)))  float  f32x8;

union FragBF16 { bf16x16 v; bf16x8 h[2]; };

static __device__ __forceinline__ unsigned short f2bf(float f) {
    unsigned u = __builtin_bit_cast(unsigned, f);
    u += 0x7FFFu + ((u >> 16) & 1u);          // round-to-nearest-even
    return (unsigned short)(u >> 16);
}

// Cubic B-spline bases on uniform grid g[i] = 0.4*i - 2.2, i = 0..11.
static __device__ __forceinline__ void bsplines8(float x, float* out) {
    float b[11];
#pragma unroll
    for (int t = 0; t < 11; ++t) {
        float g0 = 0.4f * t - 2.2f;
        b[t] = (x >= g0 && x < g0 + 0.4f) ? 1.0f : 0.0f;
    }
#pragma unroll
    for (int k = 1; k <= 3; ++k) {
        float inv = 1.0f / (0.4f * (float)k);
#pragma unroll
        for (int t = 0; t < 11 - k; ++t) {
            float gt  = 0.4f * (float)t - 2.2f;
            float gtk = 0.4f * (float)(t + k + 1) - 2.2f;
            b[t] = (x - gt) * inv * b[t] + (gtk - x) * inv * b[t + 1];
        }
    }
#pragma unroll
    for (int t = 0; t < 8; ++t) out[t] = b[t];
}

// ---------------------------------------------------------------------------
__global__ void kan_pack(const float* __restrict__ bw, const float* __restrict__ sw,
                         const float* __restrict__ sc, unsigned short* __restrict__ Wb,
                         int O, int F, int Fp) {
    int idx = blockIdx.x * blockDim.x + threadIdx.x;
    if (idx >= O * Fp) return;
    int o = idx / Fp;
    int f = idx - o * Fp;
    unsigned short* dst = Wb + (size_t)idx * 9;
    if (f >= F) {
#pragma unroll
        for (int j = 0; j < 9; ++j) dst[j] = 0;
        return;
    }
    size_t wi = (size_t)o * F + f;
    dst[0] = f2bf(bw[wi]);
    float s = sc[wi];
#pragma unroll
    for (int j = 0; j < 8; ++j) dst[1 + j] = f2bf(sw[wi * 8 + j] * s);
}

__global__ void kan_aug0(const float* __restrict__ xs, const float* __restrict__ ys,
                         unsigned short* __restrict__ X) {
    int idx = blockIdx.x * blockDim.x + threadIdx.x;
    if (idx >= BATCH * F0P) return;
    int n = idx >> 10;
    int f = idx & (F0P - 1);
    unsigned short* dst = X + (size_t)idx * 9;
    if (f >= F0) {
#pragma unroll
        for (int j = 0; j < 9; ++j) dst[j] = 0;
        return;
    }
    float v;
    if (f == 1022)      v = xs[n * POINTS + 511];
    else if (f & 1)     v = ys[n * POINTS + (f >> 1)];
    else                v = xs[n * POINTS + (f >> 1)];
    float o9[9];
    o9[0] = v > 0.0f ? v : 0.0f;
    bsplines8(v, o9 + 1);
#pragma unroll
    for (int j = 0; j < 9; ++j) dst[j] = f2bf(o9[j]);
}

__global__ void kan_augh(const float* __restrict__ H, unsigned short* __restrict__ X,
                         int total) {
    int idx = blockIdx.x * blockDim.x + threadIdx.x;
    if (idx >= total) return;
    float v = H[idx];
    unsigned short* dst = X + (size_t)idx * 9;
    float o9[9];
    o9[0] = v > 0.0f ? v : 0.0f;
    bsplines8(v, o9 + 1);
#pragma unroll
    for (int j = 0; j < 9; ++j) dst[j] = f2bf(o9[j]);
}

// ---------------------------------------------------------------------------
// C[M][N] = A[M][K] * B[N][K]^T, bf16 -> f32 WMMA.
// Block 128(M) x 256(N) x 32(K), 256 threads = 8 waves (2 M x 4 N), each wave
// a 64x64 tile = 4x4 WMMA fragments. Triple-buffered LDS, async copies.
// ---------------------------------------------------------------------------
#define ASYNC_CP16(ldsoff, gptr)                                              \
    asm volatile("global_load_async_to_lds_b128 %0, %1, off"                  \
                 :: "v"(ldsoff), "v"(gptr) : "memory")

__global__ __launch_bounds__(256) void kan_gemm_bf16(
    const unsigned short* __restrict__ A,
    const unsigned short* __restrict__ B,
    float* __restrict__ C,
    int M, int N, int K) {
    __shared__ unsigned short lA[3][128 * 32];   // 3 x 8 KB
    __shared__ unsigned short lB[3][256 * 32];   // 3 x 16 KB

    const int tid  = threadIdx.x;
    const int lane = tid & 31;
    const int wave = tid >> 5;
    const int wm   = wave & 1;    // 64-row slab
    const int wn   = wave >> 1;   // 64-col slab

    const int bn0 = blockIdx.x * 256;
    const int bm0 = blockIdx.y * 128;

    // --- async copy pattern: thread -> 16B chunk (row = tid>>2 (+64t), col8) ---
    const int crow = tid >> 2;
    const int ccol = (tid & 3) * 8;
    const unsigned short* gA0 = A + (size_t)(bm0 + crow) * K + ccol;
    const unsigned short* gA1 = gA0 + (size_t)64 * K;
    const unsigned short* gB0 = B + (size_t)(bn0 + crow) * K + ccol;
    const unsigned short* gB1 = gB0 + (size_t)64 * K;
    const unsigned short* gB2 = gB1 + (size_t)64 * K;
    const unsigned short* gB3 = gB2 + (size_t)64 * K;
    const unsigned laoff = (unsigned)(size_t)&lA[0][crow * 32 + ccol];
    const unsigned lboff = (unsigned)(size_t)&lB[0][crow * 32 + ccol];

    f32x8 acc[4][4];
    const f32x8 zero = {0.f, 0.f, 0.f, 0.f, 0.f, 0.f, 0.f, 0.f};
#pragma unroll
    for (int i = 0; i < 4; ++i)
#pragma unroll
        for (int j = 0; j < 4; ++j) acc[i][j] = zero;

    // fragment addressing (ISA 16-bit A/B VGPR layouts)
    const int arow = wm * 64 + (lane & 15);
    const int akb  = (lane & 16) ? 8 : 0;     // A: K {0..7,16..23} / {8..15,24..31}
    const int bcol = wn * 64 + (lane & 15);
    const int bkb  = (lane & 16) ? 8 : 0;     // B: K {0..7,16..23} / {8..15,24..31}

    // prologue: stages 0 (kk=0) and 1 (kk=32); 12 async ops in flight
#pragma unroll
    for (int s = 0; s < 2; ++s) {
        const int kk = s * 32;
        ASYNC_CP16(laoff + s * 8192,          gA0 + kk);
        ASYNC_CP16(laoff + s * 8192 + 4096,   gA1 + kk);
        ASYNC_CP16(lboff + s * 16384,         gB0 + kk);
        ASYNC_CP16(lboff + s * 16384 + 4096,  gB1 + kk);
        ASYNC_CP16(lboff + s * 16384 + 8192,  gB2 + kk);
        ASYNC_CP16(lboff + s * 16384 + 12288, gB3 + kk);
    }

    const int nsteps = K >> 5;
    int s = 0;       // stage being computed
    int s2 = 2;      // stage being filled
    for (int it = 0; it < nsteps; ++it) {
        // oldest 6 async copies (stage s) complete; 6 (stage s+1) may remain
        asm volatile("s_wait_asynccnt 0x6" ::: "memory");
        __syncthreads();

        // issue next stage (dummy kk=0 near the end keeps ASYNCcnt uniform)
        {
            int kn = (it + 2) * 32;
            if (kn >= K) kn = 0;
            ASYNC_CP16(laoff + s2 * 8192,          gA0 + kn);
            ASYNC_CP16(laoff + s2 * 8192 + 4096,   gA1 + kn);
            ASYNC_CP16(lboff + s2 * 16384,         gB0 + kn);
            ASYNC_CP16(lboff + s2 * 16384 + 4096,  gB1 + kn);
            ASYNC_CP16(lboff + s2 * 16384 + 8192,  gB2 + kn);
            ASYNC_CP16(lboff + s2 * 16384 + 12288, gB3 + kn);
        }

        FragBF16 bfrag[4];
#pragma unroll
        for (int nf = 0; nf < 4; ++nf) {
            const unsigned short* p = &lB[s][(bcol + nf * 16) * 32 + bkb];
            bfrag[nf].h[0] = *(const bf16x8*)(p);
            bfrag[nf].h[1] = *(const bf16x8*)(p + 16);
        }
#pragma unroll
        for (int mf = 0; mf < 4; ++mf) {
            FragBF16 afrag;
            const unsigned short* p = &lA[s][(arow + mf * 16) * 32 + akb];
            afrag.h[0] = *(const bf16x8*)(p);
            afrag.h[1] = *(const bf16x8*)(p + 16);
#pragma unroll
            for (int nf = 0; nf < 4; ++nf) {
                acc[mf][nf] = __builtin_amdgcn_wmma_f32_16x16x32_bf16(
                    false, afrag.v, false, bfrag[nf].v,
                    (short)0, acc[mf][nf], false, false);
            }
        }

        s  = (s  == 2) ? 0 : s + 1;
        s2 = (s2 == 2) ? 0 : s2 + 1;
    }

    // f32 C layout: reg r, lanes 0-15 -> M=r, lanes 16-31 -> M=8+r; N = lane&15
    const int rbase = bm0 + wm * 64 + ((lane & 16) ? 8 : 0);
    const int cc0   = bn0 + wn * 64 + (lane & 15);
#pragma unroll
    for (int mf = 0; mf < 4; ++mf)
#pragma unroll
        for (int nf = 0; nf < 4; ++nf)
#pragma unroll
            for (int r = 0; r < 8; ++r)
                C[(size_t)(rbase + mf * 16 + r) * N + (cc0 + nf * 16)] = acc[mf][nf][r];
}

// ---------------------------------------------------------------------------
extern "C" void kernel_launch(void* const* d_in, const int* in_sizes, int n_in,
                              void* d_out, int out_size, void* d_ws, size_t ws_size,
                              hipStream_t stream) {
    (void)in_sizes; (void)n_in; (void)out_size; (void)ws_size;

    const float* xs  = (const float*)d_in[0];
    const float* ys  = (const float*)d_in[1];
    const float* bw0 = (const float*)d_in[2];
    const float* sw0 = (const float*)d_in[3];
    const float* sc0 = (const float*)d_in[4];
    const float* bw1 = (const float*)d_in[5];
    const float* sw1 = (const float*)d_in[6];
    const float* sc1 = (const float*)d_in[7];
    const float* bw2 = (const float*)d_in[8];
    const float* sw2 = (const float*)d_in[9];
    const float* sc2 = (const float*)d_in[10];

    char* ws = (char*)d_ws;
    size_t off = 0;
    unsigned short* wb0 = (unsigned short*)(ws + off); off += (size_t)EMBD   * K0 * 2;
    unsigned short* wb1 = (unsigned short*)(ws + off); off += (size_t)EMBD   * K1 * 2;
    unsigned short* wb2 = (unsigned short*)(ws + off); off += (size_t)POINTS * K1 * 2;
    unsigned short* x0  = (unsigned short*)(ws + off); off += (size_t)BATCH  * K0 * 2;
    unsigned short* x1  = (unsigned short*)(ws + off); off += (size_t)BATCH  * K1 * 2;
    float*          hbuf = (float*)(ws + off);         off += (size_t)BATCH  * EMBD * 4;

    const int T = 256;
    kan_pack<<<(EMBD * F0P + T - 1) / T, T, 0, stream>>>(bw0, sw0, sc0, wb0, EMBD, F0, F0P);
    kan_pack<<<(EMBD * EMBD + T - 1) / T, T, 0, stream>>>(bw1, sw1, sc1, wb1, EMBD, EMBD, EMBD);
    kan_pack<<<(POINTS * EMBD + T - 1) / T, T, 0, stream>>>(bw2, sw2, sc2, wb2, POINTS, EMBD, EMBD);

    kan_aug0<<<(BATCH * F0P + T - 1) / T, T, 0, stream>>>(xs, ys, x0);

    kan_gemm_bf16<<<dim3(EMBD / 256, BATCH / 128), T, 0, stream>>>(x0, wb0, hbuf, BATCH, EMBD, K0);

    kan_augh<<<(BATCH * EMBD + T - 1) / T, T, 0, stream>>>(hbuf, x1, BATCH * EMBD);
    kan_gemm_bf16<<<dim3(EMBD / 256, BATCH / 128), T, 0, stream>>>(x1, wb1, hbuf, BATCH, EMBD, K1);

    kan_augh<<<(BATCH * EMBD + T - 1) / T, T, 0, stream>>>(hbuf, x1, BATCH * EMBD);
    kan_gemm_bf16<<<dim3(POINTS / 256, BATCH / 128), T, 0, stream>>>(x1, wb2, (float*)d_out,
                                                                     BATCH, POINTS, K1);
}
